// CascadedNet_52853867545072
// MI455X (gfx1250) — matrix operand order repaired and checked by reading, not compiled, not run on hardware
//
#include <hip/hip_runtime.h>
#include <hip/hip_bf16.h>
#include <math.h>

#define B_BATCH 8192

typedef __attribute__((ext_vector_type(16))) __bf16 v16bf;
typedef __attribute__((ext_vector_type(8)))  float  v8f;

union FragB16 {
  v16bf v;
  uint4 q[2];
  unsigned int u[8];
};

// float -> bf16 (round to nearest even), stored as raw u16
__device__ __forceinline__ unsigned short f2bf(float f) {
  unsigned int x = __float_as_uint(f);
  x += 0x7FFFu + ((x >> 16) & 1u);
  return (unsigned short)(x >> 16);
}

// A fragment (16x32 bf16, M x K). p = &A[m*lda + k0 + 8*half], 16B aligned.
// ISA layout: lane half h, VGPR i -> K = (i<4 ? 2i : 8+2i) + 8h  => elems p[0..7], p[16..23]
__device__ __forceinline__ void load_frag_a(FragB16& f, const unsigned short* p) {
  f.q[0] = *(const uint4*)(p);
  f.q[1] = *(const uint4*)(p + 16);
}

// B fragment (32x16 bf16, K x N), weights stored row-major over K (Bt[n][k]).
// p = &Bt[n*ldb + k0 + 8*half]; same K layout as A => elems p[0..7], p[16..23].
__device__ __forceinline__ void load_frag_b(FragB16& f, const unsigned short* p) {
  f.q[0] = *(const uint4*)(p);
  f.q[1] = *(const uint4*)(p + 16);
}

// ---------------------------------------------------------------------------
// K1: adaptive max-pool (per-batch bins) + hand-head gathers, batch reversed.
// One block per source batch element; reads res_map once.
// ---------------------------------------------------------------------------
__global__ void pool_gather_kernel(const float* __restrict__ res_1,
                                   const float* __restrict__ res_2,
                                   const float* __restrict__ res_3,
                                   const float* __restrict__ res_map,
                                   unsigned short* __restrict__ map5,  // [B][2048] bf16
                                   unsigned short* __restrict__ g)     // [4][B][128] bf16
{
  const int sb = blockIdx.x;
  const int db = B_BATCH - 1 - sb;     // batch reversal
  const int tid = threadIdx.x;

  const float a0 = res_2[sb * 2 + 0], a1 = res_2[sb * 2 + 1];
  const float c0 = res_3[sb * 2 + 0], c1 = res_3[sb * 2 + 1];
  const int l  = (int)(9.0f * fmaxf(0.0f, a0 - c0));
  const int r  = (int)(9.0f * fminf(1.0f, fabsf(a0 + c0)));
  const int up = (int)(9.0f * fmaxf(0.0f, a1 - c1));
  const int dn = (int)(9.0f * fminf(1.0f, fabsf(a1 + c1)));
  const int Hc = max(r - l, 1);
  const int Wc = max(dn - up, 1);

  const float* m = res_map + (size_t)sb * 12800;

  for (int e = tid; e < 2048; e += 256) {
    const int ic = e >> 4, i = (e >> 2) & 3, j = e & 3;
    int xs = l  + (i * Hc) / 4;
    int xe = l  + ((i + 1) * Hc + 3) / 4;
    int ys = up + (j * Wc) / 4;
    int ye = up + ((j + 1) * Wc + 3) / 4;
    if (xe > 10) xe = 10;
    if (ye > 10) ye = 10;
    float acc = -INFINITY;
    for (int x = xs; x < xe; ++x)
      for (int y = ys; y < ye; ++y)
        acc = fmaxf(acc, m[ic * 100 + x * 10 + y]);
    map5[(size_t)db * 2048 + e] = f2bf(acc);   // layout ic*16 + ih*4 + iw
  }

  for (int t = tid; t < 512; t += 256) {
    const int hk = t >> 7, ic = t & 127;
    const int cx = (int)(res_1[sb * 8 + 2 * hk] * 9.0f);
    const int cy = (int)(res_1[sb * 8 + 2 * hk + 1] * 9.0f);
    g[((size_t)hk * B_BATCH + db) * 128 + ic] = f2bf(m[ic * 100 + cx * 10 + cy]);
  }
}

// ---------------------------------------------------------------------------
// K2: weight conversion/packing to bf16 (Bt[n][k] row-major-over-K) + BN fold.
// ---------------------------------------------------------------------------
__global__ void prep_kernel(const float* __restrict__ h_w1,    // (4,64,128)
                            const float* __restrict__ conv1_w, // (256,128,4,4)
                            const float* __restrict__ conv2_w, // (512,256,4,4)
                            const float* __restrict__ fc_w,    // (54,512)
                            const float* __restrict__ bn_gamma,
                            const float* __restrict__ bn_beta,
                            const float* __restrict__ bn_mean,
                            const float* __restrict__ bn_var,
                            unsigned short* __restrict__ w1b,  // [4*64][128]
                            unsigned short* __restrict__ w1c,  // [256][2048]
                            unsigned short* __restrict__ w2c,  // [512][1024]
                            unsigned short* __restrict__ wfc,  // [64][512] padded
                            float* __restrict__ bnscale,
                            float* __restrict__ bnshift)
{
  const int t = blockIdx.x * blockDim.x + threadIdx.x;
  if (t < 4 * 64 * 128) w1b[t] = f2bf(h_w1[t]);
  if (t < 256 * 2048)   w1c[t] = f2bf(conv1_w[t]); // k = ic*16+kh*4+kw already
  if (t < 512 * 1024) {
    const int oc = t >> 10, k = t & 1023;
    const int ic = k >> 2, ih = (k >> 1) & 1, iw = k & 1;
    // only kernel taps (kh,kw) in {1,2} hit valid input for the single output
    w2c[t] = f2bf(conv2_w[oc * 4096 + ic * 16 + (ih + 1) * 4 + (iw + 1)]);
  }
  if (t < 64 * 512) {
    const int oc = t >> 9, k = t & 511;
    wfc[t] = (oc < 54) ? f2bf(fc_w[oc * 512 + k]) : (unsigned short)0;
  }
  if (t < 256) {
    const float s = bn_gamma[t] * rsqrtf(bn_var[t] + 1e-5f);
    bnscale[t] = s;
    bnshift[t] = bn_beta[t] - bn_mean[t] * s;
  }
}

// ---------------------------------------------------------------------------
// K3: hand-head layer1: H = leaky_relu(g @ w1^T + b1), WMMA bf16 GEMM
// grid: (B/16, 4 heads), block 128 (4 waves x 16 N-cols)
// ---------------------------------------------------------------------------
__global__ void hand_l1_kernel(const unsigned short* __restrict__ g,   // [4][B][128]
                               const unsigned short* __restrict__ w1,  // [4*64][128]
                               const float* __restrict__ b1,           // (4,64)
                               float* __restrict__ H)                  // [4][B][64]
{
  const int hk   = blockIdx.y;
  const int b0   = blockIdx.x * 16;
  const int wave = threadIdx.x >> 5;
  const int lane = threadIdx.x & 31;
  const int half = lane >> 4;
  const int mr   = lane & 15;

  const unsigned short* Abase = g  + ((size_t)hk * B_BATCH + b0 + mr) * 128 + 8 * half;
  const unsigned short* Bbase = w1 + ((size_t)(hk * 64 + wave * 16 + mr)) * 128 + 8 * half;

  v8f acc = {};
#pragma unroll
  for (int k0 = 0; k0 < 128; k0 += 32) {
    FragB16 a, b;
    load_frag_a(a, Abase + k0);
    load_frag_b(b, Bbase + k0);
    acc = __builtin_amdgcn_wmma_f32_16x16x32_bf16(false, a.v, false, b.v,
                                                  (short)0, acc, false, false);
  }
  const int n = wave * 16 + mr;
  const float bias = b1[hk * 64 + n];
#pragma unroll
  for (int rr = 0; rr < 8; ++rr) {
    const int mrow = rr + 8 * half;
    float v = acc[rr] + bias;
    v = (v > 0.0f) ? v : 0.2f * v;
    H[((size_t)hk * B_BATCH + b0 + mrow) * 64 + n] = v;
  }
}

// ---------------------------------------------------------------------------
// K4: hand-head layer2 (N=2, scalar): ps = sigmoid(H @ w2^T + b2)
// ---------------------------------------------------------------------------
__global__ void hand_l2_kernel(const float* __restrict__ H,
                               const float* __restrict__ h_w2,  // (4,2,64)
                               const float* __restrict__ h_b2,  // (4,2)
                               float* __restrict__ out)         // d_out[0 .. 8B)
{
  const int t = blockIdx.x * blockDim.x + threadIdx.x;
  if (t >= 4 * B_BATCH * 2) return;
  const int hk  = t / (B_BATCH * 2);
  const int rem = t - hk * (B_BATCH * 2);
  const int b = rem >> 1, o = rem & 1;
  const float* h = H + ((size_t)hk * B_BATCH + b) * 64;
  const float* w = h_w2 + (hk * 2 + o) * 64;
  float s = h_b2[hk * 2 + o];
#pragma unroll
  for (int j = 0; j < 64; ++j) s += h[j] * w[j];
  out[(size_t)hk * B_BATCH * 2 + rem] = 1.0f / (1.0f + expf(-s));
}

// ---------------------------------------------------------------------------
// K5: conv1 as 4 position-GEMMs (B x 2048)(2048 x 256), zero-padded A via LDS,
// fused BN + leakyReLU. grid (B/16, 256/64, 4 positions), block 128.
// ---------------------------------------------------------------------------
__global__ void conv1_kernel(const unsigned short* __restrict__ map5, // [B][2048]
                             const unsigned short* __restrict__ w,    // [256][2048]
                             const float* __restrict__ bnscale,
                             const float* __restrict__ bnshift,
                             unsigned short* __restrict__ y1)         // [B][1024] (oc*4+p)
{
  __shared__ __attribute__((aligned(16))) unsigned short At[16 * 32];

  const int b0  = blockIdx.x * 16;
  const int ocg = blockIdx.y;          // 0..3
  const int p   = blockIdx.z;          // 0..3
  const int oy = p >> 1, ox = p & 1;
  const int tid  = threadIdx.x;
  const int wave = tid >> 5;
  const int lane = tid & 31;
  const int half = lane >> 4;
  const int mr   = lane & 15;

  const unsigned short* Bbase = w + ((size_t)(ocg * 64 + wave * 16 + mr)) * 2048 + 8 * half;
  const unsigned short* Albase = At + mr * 32 + 8 * half;

  v8f acc = {};
  for (int k0 = 0; k0 < 2048; k0 += 32) {
    // cooperative gather of zero-padded A tile (16 rows x 32 k) into LDS
#pragma unroll
    for (int tt = 0; tt < 4; ++tt) {
      const int e   = tt * 128 + tid;     // 0..511
      const int row = e >> 5;             // batch row in tile
      const int k   = k0 + (e & 31);
      const int ic  = k >> 4;
      const int kh  = (k >> 2) & 3;
      const int kw  = k & 3;
      const int ih  = 2 * oy + kh - 1;
      const int iw  = 2 * ox + kw - 1;
      unsigned short v = 0;
      if ((unsigned)ih < 4u && (unsigned)iw < 4u)
        v = map5[(size_t)(b0 + row) * 2048 + ic * 16 + ih * 4 + iw];
      At[e] = v;
    }
    __syncthreads();
    FragB16 a, b;
    load_frag_a(a, Albase);              // ds_load_b128 x2
    load_frag_b(b, Bbase + k0);
    acc = __builtin_amdgcn_wmma_f32_16x16x32_bf16(false, a.v, false, b.v,
                                                  (short)0, acc, false, false);
    __syncthreads();
  }
  const int oc = ocg * 64 + wave * 16 + mr;
  const float s = bnscale[oc], sh = bnshift[oc];
#pragma unroll
  for (int rr = 0; rr < 8; ++rr) {
    const int mrow = rr + 8 * half;
    float v = acc[rr] * s + sh;
    v = (v > 0.0f) ? v : 0.2f * v;
    y1[(size_t)(b0 + mrow) * 1024 + oc * 4 + p] = f2bf(v);
  }
}

// ---------------------------------------------------------------------------
// K6: conv2 as GEMM (B x 1024)(1024 x 512). grid (B/16, 8), block 128.
// ---------------------------------------------------------------------------
__global__ void conv2_kernel(const unsigned short* __restrict__ y1,  // [B][1024]
                             const unsigned short* __restrict__ w,   // [512][1024]
                             unsigned short* __restrict__ y2)        // [B][512]
{
  const int b0  = blockIdx.x * 16;
  const int ocg = blockIdx.y;          // 0..7
  const int wave = threadIdx.x >> 5;
  const int lane = threadIdx.x & 31;
  const int half = lane >> 4;
  const int mr   = lane & 15;

  const unsigned short* Abase = y1 + (size_t)(b0 + mr) * 1024 + 8 * half;
  const unsigned short* Bbase = w + ((size_t)(ocg * 64 + wave * 16 + mr)) * 1024 + 8 * half;

  v8f acc = {};
  for (int k0 = 0; k0 < 1024; k0 += 32) {
    FragB16 a, b;
    load_frag_a(a, Abase + k0);
    load_frag_b(b, Bbase + k0);
    acc = __builtin_amdgcn_wmma_f32_16x16x32_bf16(false, a.v, false, b.v,
                                                  (short)0, acc, false, false);
  }
  const int oc = ocg * 64 + wave * 16 + mr;
#pragma unroll
  for (int rr = 0; rr < 8; ++rr) {
    const int mrow = rr + 8 * half;
    y2[(size_t)(b0 + mrow) * 512 + oc] = f2bf(acc[rr]);
  }
}

// ---------------------------------------------------------------------------
// K7: FC (B x 512)(512 x 64pad) + bias + sigmoid. grid (B/16), block 128.
// ---------------------------------------------------------------------------
__global__ void fc_kernel(const unsigned short* __restrict__ y2,  // [B][512]
                          const unsigned short* __restrict__ w,   // [64][512]
                          const float* __restrict__ fc_b,         // (54,)
                          float* __restrict__ out)                // d_out + 8*B, [B][54]
{
  const int b0   = blockIdx.x * 16;
  const int wave = threadIdx.x >> 5;
  const int lane = threadIdx.x & 31;
  const int half = lane >> 4;
  const int mr   = lane & 15;

  const unsigned short* Abase = y2 + (size_t)(b0 + mr) * 512 + 8 * half;
  const unsigned short* Bbase = w + ((size_t)(wave * 16 + mr)) * 512 + 8 * half;

  v8f acc = {};
#pragma unroll
  for (int k0 = 0; k0 < 512; k0 += 32) {
    FragB16 a, b;
    load_frag_a(a, Abase + k0);
    load_frag_b(b, Bbase + k0);
    acc = __builtin_amdgcn_wmma_f32_16x16x32_bf16(false, a.v, false, b.v,
                                                  (short)0, acc, false, false);
  }
  const int n = wave * 16 + mr;
  if (n < 54) {
    const float bias = fc_b[n];
#pragma unroll
    for (int rr = 0; rr < 8; ++rr) {
      const int mrow = rr + 8 * half;
      const float v = acc[rr] + bias;
      out[(size_t)(b0 + mrow) * 54 + n] = 1.0f / (1.0f + expf(-v));
    }
  }
}

// ---------------------------------------------------------------------------
extern "C" void kernel_launch(void* const* d_in, const int* in_sizes, int n_in,
                              void* d_out, int out_size, void* d_ws, size_t ws_size,
                              hipStream_t stream) {
  (void)in_sizes; (void)n_in; (void)out_size; (void)ws_size;

  const float* res_1    = (const float*)d_in[0];
  const float* res_2    = (const float*)d_in[1];
  const float* res_3    = (const float*)d_in[2];
  const float* res_map  = (const float*)d_in[3];
  const float* h_w1     = (const float*)d_in[4];
  const float* h_b1     = (const float*)d_in[5];
  const float* h_w2     = (const float*)d_in[6];
  const float* h_b2     = (const float*)d_in[7];
  const float* conv1_w  = (const float*)d_in[8];
  const float* bn_gamma = (const float*)d_in[9];
  const float* bn_beta  = (const float*)d_in[10];
  const float* bn_mean  = (const float*)d_in[11];
  const float* bn_var   = (const float*)d_in[12];
  const float* conv2_w  = (const float*)d_in[13];
  const float* fc_w     = (const float*)d_in[14];
  const float* fc_b     = (const float*)d_in[15];
  float* out = (float*)d_out;

  char* ws = (char*)d_ws;
  size_t off = 0;
  auto take = [&](size_t bytes) -> char* {
    char* p = ws + off;
    off = (off + bytes + 255) & ~(size_t)255;
    return p;
  };
  unsigned short* g_ws    = (unsigned short*)take((size_t)4 * B_BATCH * 128 * 2);
  unsigned short* map5_ws = (unsigned short*)take((size_t)B_BATCH * 2048 * 2);
  float*          H_ws    = (float*)take((size_t)4 * B_BATCH * 64 * 4);
  unsigned short* y1_ws   = (unsigned short*)take((size_t)B_BATCH * 1024 * 2);
  unsigned short* y2_ws   = (unsigned short*)take((size_t)B_BATCH * 512 * 2);
  unsigned short* w1b_ws  = (unsigned short*)take((size_t)4 * 64 * 128 * 2);
  unsigned short* w1c_ws  = (unsigned short*)take((size_t)256 * 2048 * 2);
  unsigned short* w2c_ws  = (unsigned short*)take((size_t)512 * 1024 * 2);
  unsigned short* wfc_ws  = (unsigned short*)take((size_t)64 * 512 * 2);
  float*          bns_ws  = (float*)take((size_t)256 * 4);
  float*          bnh_ws  = (float*)take((size_t)256 * 4);

  // independent prep stages
  prep_kernel<<<(512 * 1024 + 255) / 256, 256, 0, stream>>>(
      h_w1, conv1_w, conv2_w, fc_w, bn_gamma, bn_beta, bn_mean, bn_var,
      w1b_ws, w1c_ws, w2c_ws, wfc_ws, bns_ws, bnh_ws);
  pool_gather_kernel<<<B_BATCH, 256, 0, stream>>>(
      res_1, res_2, res_3, res_map, map5_ws, g_ws);

  // hand heads
  hand_l1_kernel<<<dim3(B_BATCH / 16, 4), 128, 0, stream>>>(g_ws, w1b_ws, h_b1, H_ws);
  hand_l2_kernel<<<(4 * B_BATCH * 2 + 255) / 256, 256, 0, stream>>>(H_ws, h_w2, h_b2, out);

  // conv tower
  conv1_kernel<<<dim3(B_BATCH / 16, 4, 4), 128, 0, stream>>>(
      map5_ws, w1c_ws, bns_ws, bnh_ws, y1_ws);
  conv2_kernel<<<dim3(B_BATCH / 16, 8), 128, 0, stream>>>(y1_ws, w2c_ws, y2_ws);
  fc_kernel<<<B_BATCH / 16, 128, 0, stream>>>(y2_ws, wfc_ws, fc_b,
                                              out + (size_t)8 * B_BATCH);
}
